// SimplifiedSSMLayer_51144470561385
// MI455X (gfx1250) — matrix operand order repaired and checked by reading, compiled, not run on hardware
//
#include <hip/hip_runtime.h>
#include <math.h>

// ---------------- types ----------------
typedef __bf16 bf16_t;
typedef bf16_t v16bf __attribute__((ext_vector_type(16)));
typedef float  v8f   __attribute__((ext_vector_type(8)));
typedef float  v4f   __attribute__((ext_vector_type(4)));

#define DM    1024
#define DS    16
#define SEQ   4096
#define BATCH 8
#define NTOK  (BATCH * SEQ)   // 32768

// ---------------- ws layout (bytes) ----------------
// X16 : NTOK*DM bf16   = 64 MiB
// GW16: DM*DM  bf16    =  2 MiB
// BW16: DS*DM  bf16    = 32 KiB
// BX  : NTOK*DS f32    =  2 MiB
// ST  : NTOK*DS f32    =  2 MiB
#define OFF_X16  ((size_t)0)
#define OFF_GW16 ((size_t)67108864)
#define OFF_BW16 ((size_t)69206016)
#define OFF_BX   ((size_t)69238784)
#define OFF_ST   ((size_t)71335936)

// ---------------- kernel 1: f32 -> bf16 conversion ----------------
__global__ void k_convert(const float* __restrict__ x,
                          const float* __restrict__ gw,
                          const float* __restrict__ bw,
                          bf16_t* __restrict__ x16,
                          bf16_t* __restrict__ gw16,
                          bf16_t* __restrict__ bw16) {
    int stride = gridDim.x * blockDim.x;
    int t0 = blockIdx.x * blockDim.x + threadIdx.x;
    for (int i = t0; i < NTOK * DM; i += stride) x16[i]  = (bf16_t)x[i];
    for (int i = t0; i < DM * DM;   i += stride) gw16[i] = (bf16_t)gw[i];
    for (int i = t0; i < DS * DM;   i += stride) bw16[i] = (bf16_t)bw[i];
}

// ---------------- kernel 2: Bx = x @ B_w^T + B_b  ([NTOK,16]) ----------------
// One wave per 16-row tile. WMMA bf16 16x16x32, K = 1024.
__global__ void k_bx(const bf16_t* __restrict__ x16,
                     const bf16_t* __restrict__ bw16,
                     const float*  __restrict__ bb,
                     float* __restrict__ Bx) {
    int wave = (int)((blockIdx.x * blockDim.x + threadIdx.x) >> 5);
    int lane = threadIdx.x & 31;
    int l15  = lane & 15;
    int hi   = lane >> 4;
    int rowbase = wave * 16;
    if (rowbase >= NTOK) return;

    const bf16_t* arow = x16  + (size_t)(rowbase + l15) * DM;
    const bf16_t* brow = bw16 + (size_t)l15 * DM;

    v8f acc = {};
    for (int k0 = 0; k0 < DM; k0 += 32) {
        v16bf A, B;
        int ka = k0 + hi * 8;   // A: lane<16 -> K {0..7,16..23}; lane>=16 -> {8..15,24..31}
        reinterpret_cast<v4f*>(&A)[0] = *reinterpret_cast<const v4f*>(arow + ka);
        reinterpret_cast<v4f*>(&A)[1] = *reinterpret_cast<const v4f*>(arow + ka + 16);
        int kb = k0 + hi * 8;   // B: lane<16 -> K {0..7,16..23}; lane>=16 -> {8..15,24..31}
        reinterpret_cast<v4f*>(&B)[0] = *reinterpret_cast<const v4f*>(brow + kb);
        reinterpret_cast<v4f*>(&B)[1] = *reinterpret_cast<const v4f*>(brow + kb + 16);
        acc = __builtin_amdgcn_wmma_f32_16x16x32_bf16(false, A, false, B,
                                                      (short)0, acc, false, false);
    }
    float bbn = bb[l15];
#pragma unroll
    for (int r = 0; r < 8; ++r) {
        int m = rowbase + r + hi * 8;       // C/D layout: VGPR r -> M=r (lanes<16), M=r+8 (lanes>=16)
        Bx[(size_t)m * DS + l15] = acc[r] + bbn;
    }
}

// ---------------- kernel 3: sequential scan ----------------
// 128 threads: 8 batches x 16 state-rows. state_n = tanh(sum_m A[n][m]*state_m + bx_n)
__global__ void k_scan(const float* __restrict__ Amat,
                       const float* __restrict__ Bx,
                       float* __restrict__ ST) {
    int tid  = threadIdx.x;         // 0..127
    int b    = tid >> 4;            // batch
    int n    = tid & 15;            // state row
    int lane = tid & 31;
    int srcbase = lane & 16;        // shuffle within own 16-lane half

    float a[16];
#pragma unroll
    for (int m = 0; m < 16; ++m) a[m] = Amat[n * 16 + m];

    const float* bxp = Bx + (size_t)b * SEQ * DS + n;
    float*       stp = ST + (size_t)b * SEQ * DS + n;

    float st = 0.0f;
    for (int t = 0; t < SEQ; ++t) {
        float acc = bxp[(size_t)t * DS];
#pragma unroll
        for (int m = 0; m < 16; ++m)
            acc = fmaf(a[m], __shfl(st, srcbase + m, 32), acc);
        st = tanhf(acc);
        stp[(size_t)t * DS] = st;
    }
}

// ---------------- kernel 4: gate GEMM + fused SSM-output epilogue ----------------
// Wave computes a 16(M) x 64(N) tile of gate = x @ gate_w^T; epilogue fuses:
//   g = silu(gate + gate_b); ssm = states@C_w^T + C_b + D*x; out_pre = g*ssm + (1-g)*x
__global__ void k_gate(const bf16_t* __restrict__ x16,
                       const bf16_t* __restrict__ gw16,
                       const float*  __restrict__ x,
                       const float*  __restrict__ ST,
                       const float*  __restrict__ Cw,
                       const float*  __restrict__ Cb,
                       const float*  __restrict__ Dd,
                       const float*  __restrict__ gb,
                       float* __restrict__ out) {
    int wid  = (int)((blockIdx.x * blockDim.x + threadIdx.x) >> 5);
    int lane = threadIdx.x & 31;
    int l15  = lane & 15;
    int hi   = lane >> 4;
    int mbase = (wid >> 4) * 16;    // 2048 M-tiles
    int nbase = (wid & 15) * 64;    // 16 N-tiles of 64

    const bf16_t* arow = x16  + (size_t)(mbase + l15) * DM;
    const bf16_t* brow = gw16 + (size_t)(nbase + l15) * DM;

    v8f acc[4] = {{}, {}, {}, {}};
    for (int k0 = 0; k0 < DM; k0 += 32) {
        v16bf A;
        int ka = k0 + hi * 8;
        reinterpret_cast<v4f*>(&A)[0] = *reinterpret_cast<const v4f*>(arow + ka);
        reinterpret_cast<v4f*>(&A)[1] = *reinterpret_cast<const v4f*>(arow + ka + 16);
        int kb = k0 + hi * 8;
#pragma unroll
        for (int j = 0; j < 4; ++j) {
            v16bf B;
            const bf16_t* bp = brow + (size_t)j * 16 * DM + kb;
            reinterpret_cast<v4f*>(&B)[0] = *reinterpret_cast<const v4f*>(bp);
            reinterpret_cast<v4f*>(&B)[1] = *reinterpret_cast<const v4f*>(bp + 16);
            acc[j] = __builtin_amdgcn_wmma_f32_16x16x32_bf16(false, A, false, B,
                                                             (short)0, acc[j], false, false);
        }
    }

#pragma unroll
    for (int r = 0; r < 8; ++r) {
        int row = mbase + r + hi * 8;
        const float* sp = ST + (size_t)row * DS;
        float strow[16];
#pragma unroll
        for (int q = 0; q < 16; ++q) strow[q] = sp[q];
#pragma unroll
        for (int j = 0; j < 4; ++j) {
            int e = nbase + j * 16 + l15;
            float gin = acc[j][r] + gb[e];
            float g = gin / (1.0f + __expf(-gin));          // silu
            const float* cw = Cw + (size_t)e * DS;
            float ssm = Cb[e];
#pragma unroll
            for (int q = 0; q < 16; ++q) ssm = fmaf(strow[q], cw[q], ssm);
            float xv = x[(size_t)row * DM + e];
            ssm = fmaf(Dd[e], xv, ssm);
            out[(size_t)row * DM + e] = g * ssm + (1.0f - g) * xv;  // pre-LN (residual added in LN)
        }
    }
}

// ---------------- kernel 5: residual + LayerNorm (in-place on d_out) ----------------
__global__ void k_ln(const float* __restrict__ x,
                     const float* __restrict__ lng,
                     const float* __restrict__ lnb,
                     float* __restrict__ out) {
    int row = blockIdx.x;
    int tid = threadIdx.x;          // 256 threads, 4 elems each
    const float* xr = x + (size_t)row * DM;
    float* orow = out + (size_t)row * DM;

    float v[4];
    float s = 0.0f, s2 = 0.0f;
#pragma unroll
    for (int i = 0; i < 4; ++i) {
        int c = tid + i * 256;
        float t = orow[c] + xr[c];  // + residual
        v[i] = t;
        s += t;
        s2 += t * t;
    }
#pragma unroll
    for (int off = 16; off > 0; off >>= 1) {
        s  += __shfl_down(s,  off, 32);
        s2 += __shfl_down(s2, off, 32);
    }
    __shared__ float ls[8], ls2[8];
    int wid = tid >> 5, lane = tid & 31;
    if (lane == 0) { ls[wid] = s; ls2[wid] = s2; }
    __syncthreads();
    if (tid == 0) {
        float a = 0.0f, bsum = 0.0f;
#pragma unroll
        for (int i = 0; i < 8; ++i) { a += ls[i]; bsum += ls2[i]; }
        ls[0] = a; ls2[0] = bsum;
    }
    __syncthreads();
    float mu  = ls[0]  * (1.0f / DM);
    float var = ls2[0] * (1.0f / DM) - mu * mu;
    float inv = rsqrtf(var + 1e-5f);
#pragma unroll
    for (int i = 0; i < 4; ++i) {
        int c = tid + i * 256;
        orow[c] = (v[i] - mu) * inv * lng[c] + lnb[c];
    }
}

// ---------------- launch ----------------
extern "C" void kernel_launch(void* const* d_in, const int* in_sizes, int n_in,
                              void* d_out, int out_size, void* d_ws, size_t ws_size,
                              hipStream_t stream) {
    const float* x    = (const float*)d_in[0];
    const float* Amat = (const float*)d_in[1];
    const float* B_w  = (const float*)d_in[2];
    const float* B_b  = (const float*)d_in[3];
    const float* C_w  = (const float*)d_in[4];
    const float* C_b  = (const float*)d_in[5];
    const float* Dvec = (const float*)d_in[6];
    const float* g_w  = (const float*)d_in[7];
    const float* g_b  = (const float*)d_in[8];
    const float* ln_g = (const float*)d_in[9];
    const float* ln_b = (const float*)d_in[10];
    float* out = (float*)d_out;

    char* ws = (char*)d_ws;
    bf16_t* X16  = (bf16_t*)(ws + OFF_X16);
    bf16_t* GW16 = (bf16_t*)(ws + OFF_GW16);
    bf16_t* BW16 = (bf16_t*)(ws + OFF_BW16);
    float*  BX   = (float*)(ws + OFF_BX);
    float*  STt  = (float*)(ws + OFF_ST);

    k_convert<<<2048, 256, 0, stream>>>(x, g_w, B_w, X16, GW16, BW16);
    k_bx<<<NTOK / 16 / 8, 256, 0, stream>>>(X16, BW16, B_b, BX);
    k_scan<<<1, 128, 0, stream>>>(Amat, BX, STt);
    k_gate<<<(NTOK / 16) * (DM / 64) / 8, 256, 0, stream>>>(X16, GW16, x, STt,
                                                            C_w, C_b, Dvec, g_b, out);
    k_ln<<<NTOK, 256, 0, stream>>>(x, ln_g, ln_b, out);
}
